// BahdanauAttnDecoderRNN_66984309948607
// MI455X (gfx1250) — matrix operand order repaired and checked by reading, not compiled, not run on hardware
//
#include <hip/hip_runtime.h>
#include <hip/hip_bf16.h>
#include <math.h>

// Problem constants (B, T, H, D, O) = (64, 4096, 256, 128, 128)
#define Bv 64
#define Tv 4096
#define Hv 256
#define Dv 128
#define Ov 128
#define EPSv 1e-5f

typedef __bf16 bf16_t;
typedef __attribute__((ext_vector_type(16))) __bf16        v16bf;
typedef __attribute__((ext_vector_type(8)))  float         v8f;
typedef __attribute__((ext_vector_type(4)))  float         f32x4;
typedef __attribute__((ext_vector_type(4)))  unsigned int  u32x4;
typedef __attribute__((ext_vector_type(8)))  unsigned int  u32x8;

// LDS row stride for W_e staging: 256 + 8 pad bf16 elements = 528 bytes
// 528/4 = 132 dwords; 132 % 64 = 4 -> consecutive g rows start 4 banks apart,
// so the 16 lanes of a B-fragment load hit 16 disjoint 4-bank groups.
#define LDSTRIDE 264

// Persistent-kernel sizing: 2048 M-chunks of 128 rows, 512 resident blocks.
#define NCHUNK ((Tv * Bv) / 128)
#define NBLOCK 512

// ---------------------------------------------------------------------------
// small helpers
// ---------------------------------------------------------------------------
__device__ __forceinline__ float dot_n(const float* __restrict__ a,
                                       const float* __restrict__ w, int n) {
  float acc = 0.f;
  for (int k = 0; k < n; k += 4) {
    float4 x = *(const float4*)(a + k);
    float4 y = *(const float4*)(w + k);
    acc += x.x * y.x + x.y * y.y + x.z * y.z + x.w * y.w;
  }
  return acc;
}

__device__ __forceinline__ float sigmoidf_(float x) {
  return 1.0f / (1.0f + expf(-x));
}

// Branch-free tanh for the hot loop: copysign(1 - 2/(exp(2|x|)+1), x).
// exp(inf)->inf => rcp->0 => +/-1 saturation; no EXEC divergence.
__device__ __forceinline__ float fast_tanh(float x) {
  float ax = fabsf(x);
  float e  = __expf(ax + ax);                    // v_exp_f32 path
  float t  = 1.0f - 2.0f * __builtin_amdgcn_rcpf(e + 1.0f);
  return copysignf(t, x);
}

// ---------------------------------------------------------------------------
// Prep: convert W_e (= attn_W[:, H:2H]) to bf16, row-major [g][h]
// ---------------------------------------------------------------------------
__global__ void k_cvt_we(const float* __restrict__ attn_W,
                         bf16_t* __restrict__ we) {
  int idx = blockIdx.x * blockDim.x + threadIdx.x;   // 65536 threads
  int g = idx >> 8, h = idx & 255;
  we[idx] = (bf16_t)attn_W[g * (2 * Hv) + Hv + h];
}

// hid_proj[b,g] = h_prev[b,:] . W_h[g,:] + attn_b[g], stored TRANSPOSED [g][b]
__global__ void k_hidproj(const float* __restrict__ last_hidden, // [B][H]
                          const float* __restrict__ attn_W,      // [H][2H]
                          const float* __restrict__ attn_b,
                          float* __restrict__ hpT) {              // [H][B]
  int idx = blockIdx.x * blockDim.x + threadIdx.x;   // 16384 threads
  int b = idx >> 8, g = idx & 255;
  float acc = dot_n(last_hidden + (size_t)b * Hv, attn_W + (size_t)g * (2 * Hv), Hv);
  hpT[(size_t)g * Bv + b] = acc + attn_b[g];
}

// ---------------------------------------------------------------------------
// Main kernel: scores[b,t] = v . tanh(hid_proj[b,:] + enc[t,b,:] @ W_e^T)
// Persistent blocks: stage W_e in LDS once, then grid-stride over 128-row
// M-chunks (M = flat t*64+b). One wave per 16-row tile, K=256 via 8x wmma
// bf16, N=256 swept in 16 tiles with tanh/v-dot fused.
// ---------------------------------------------------------------------------
__global__ __launch_bounds__(256) void k_scores(
    const float*  __restrict__ enc,     // [T][B][H] fp32
    const bf16_t* __restrict__ we,      // [H][H] bf16 (g-major)
    const float*  __restrict__ hpT,     // [H][B]
    const float*  __restrict__ v,       // [H]
    float*        __restrict__ scores)  // [B][T]
{
  __shared__ bf16_t lds[256 * LDSTRIDE];          // 132 KB W_e stage

  const int tid = threadIdx.x;
  // Stage W_e into LDS once per resident block (dword copies, 2 bf16 each)
  for (int i = tid; i < 256 * 128; i += 256) {
    int g = i >> 7, c = i & 127;
    *(unsigned int*)&lds[g * LDSTRIDE + c * 2] = ((const unsigned int*)we)[g * 128 + c];
  }
  __syncthreads();

  const int lane  = tid & 31;
  const int wv    = tid >> 5;          // 8 waves per block
  const int sel   = lane >> 4;         // lane-half select
  const int l16   = lane & 15;

  // Preload v: lane covers column g = nt*16 + l16 for every n-tile
  float vv[16];
#pragma unroll
  for (int nt = 0; nt < 16; ++nt) vv[nt] = v[nt * 16 + l16];

  for (int chunk = blockIdx.x; chunk < NCHUNK; chunk += NBLOCK) {
    const int mBase = chunk * 128 + wv * 16;   // 16 rows per wave
    const int bBase = mBase & 63;              // rows map to b = bBase + M

    // Load + convert A fragments for this wave's 16 rows, all K=256.
    // ISA 16-bit A layout: lanes 0-15 row M=lane, K={0..7,16..23} of each
    // k32 window; lanes 16-31 same rows, K={8..15,24..31}.
    v16bf aF[8];
    {
      const float* rowp = enc + (size_t)(mBase + l16) * Hv;
#pragma unroll
      for (int kt = 0; kt < 8; ++kt) {
        const int k0 = kt * 32 + sel * 8;
        f32x4 q0 = __builtin_nontemporal_load((const f32x4*)(rowp + k0));
        f32x4 q1 = __builtin_nontemporal_load((const f32x4*)(rowp + k0 + 4));
        f32x4 q2 = __builtin_nontemporal_load((const f32x4*)(rowp + k0 + 16));
        f32x4 q3 = __builtin_nontemporal_load((const f32x4*)(rowp + k0 + 20));
        v16bf a;
        a[0]  = (bf16_t)q0[0]; a[1]  = (bf16_t)q0[1]; a[2]  = (bf16_t)q0[2]; a[3]  = (bf16_t)q0[3];
        a[4]  = (bf16_t)q1[0]; a[5]  = (bf16_t)q1[1]; a[6]  = (bf16_t)q1[2]; a[7]  = (bf16_t)q1[3];
        a[8]  = (bf16_t)q2[0]; a[9]  = (bf16_t)q2[1]; a[10] = (bf16_t)q2[2]; a[11] = (bf16_t)q2[3];
        a[12] = (bf16_t)q3[0]; a[13] = (bf16_t)q3[1]; a[14] = (bf16_t)q3[2]; a[15] = (bf16_t)q3[3];
        aF[kt] = a;
      }
    }

    float acc[8];
#pragma unroll
    for (int r = 0; r < 8; ++r) acc[r] = 0.f;

    for (int nt = 0; nt < 16; ++nt) {
      const int g = nt * 16 + l16;               // B column for this lane
      const bf16_t* bp = &lds[g * LDSTRIDE + sel * 8];

      // Issue ALL B-fragment LDS loads for this n-tile first, then the WMMA
      // chain: lets loads pipeline ahead of the matrix ops (occupancy is
      // LDS-bound, so the extra VGPRs are free).
      v16bf bF[8];
#pragma unroll
      for (int kt = 0; kt < 8; ++kt) {
        // B 16-bit layout: lane half holds K={8h..8h+7, 16+8h..16+8h+7} of col g.
        u32x4 lo = *(const u32x4*)(bp + kt * 32);
        u32x4 hi = *(const u32x4*)(bp + kt * 32 + 16);
        u32x8 cat = __builtin_shufflevector(lo, hi, 0, 1, 2, 3, 4, 5, 6, 7);
        bF[kt] = __builtin_bit_cast(v16bf, cat);
      }

      v8f c = {};
#pragma unroll
      for (int kt = 0; kt < 8; ++kt)
        c = __builtin_amdgcn_wmma_f32_16x16x32_bf16(false, aF[kt], false, bF[kt],
                                                    (short)0, c, false, false);

      // D layout: element r -> (M = r + 8*sel, N = l16). b = bBase + M.
      const float* hpp = hpT + (size_t)g * Bv + bBase + sel * 8;
      float4 h0 = *(const float4*)(hpp);
      float4 h1 = *(const float4*)(hpp + 4);
      float hp[8] = {h0.x, h0.y, h0.z, h0.w, h1.x, h1.y, h1.z, h1.w};
#pragma unroll
      for (int r = 0; r < 8; ++r)
        acc[r] += vv[nt] * fast_tanh(c[r] + hp[r]);
    }

    // Reduce partial v-dots over the 16 lanes of each half (each half owns
    // rows M=0..7 / 8..15; N covered by l16 + 16*nt).
#pragma unroll
    for (int off = 1; off < 16; off <<= 1) {
#pragma unroll
      for (int r = 0; r < 8; ++r)
        acc[r] += __shfl_xor(acc[r], off, 16);
    }

    if (l16 == 0) {
#pragma unroll
      for (int r = 0; r < 8; ++r) {
        int m = mBase + sel * 8 + r;             // m = t*64 + b
        scores[(size_t)(m & 63) * Tv + (m >> 6)] = acc[r];
      }
    }
  }
}

// ---------------------------------------------------------------------------
// Softmax over t per batch row (deterministic block reduction)
// ---------------------------------------------------------------------------
__global__ void k_softmax(const float* __restrict__ scores,
                          float* __restrict__ attn) {     // -> d_out attn slot
  const int b = blockIdx.x, tid = threadIdx.x;
  __shared__ float red[256];
  const float* row = scores + (size_t)b * Tv;
  float m = -INFINITY;
  for (int t = tid; t < Tv; t += 256) m = fmaxf(m, row[t]);
  red[tid] = m; __syncthreads();
  for (int s = 128; s > 0; s >>= 1) {
    if (tid < s) red[tid] = fmaxf(red[tid], red[tid + s]);
    __syncthreads();
  }
  const float mx = red[0]; __syncthreads();
  float sum = 0.f;
  for (int t = tid; t < Tv; t += 256) sum += expf(row[t] - mx);
  red[tid] = sum; __syncthreads();
  for (int s = 128; s > 0; s >>= 1) {
    if (tid < s) red[tid] += red[tid + s];
    __syncthreads();
  }
  const float inv = 1.0f / red[0];
  float* arow = attn + (size_t)b * Tv;
  for (int t = tid; t < Tv; t += 256) arow[t] = expf(row[t] - mx) * inv;
}

// ---------------------------------------------------------------------------
// context[b,h] = sum_t w[b,t] * enc[t,b,h]  (chunked partials, no atomics)
// ---------------------------------------------------------------------------
__global__ void k_ctx_part(const float* __restrict__ enc,
                           const float* __restrict__ attn,
                           float* __restrict__ part) {   // [32][B][H]
  const int h = threadIdx.x, chunk = blockIdx.x, b = blockIdx.y;
  float acc = 0.f;
  const int t0 = chunk * (Tv / 32);
  for (int t = t0; t < t0 + (Tv / 32); ++t) {
    float w = attn[(size_t)b * Tv + t];
    float e = __builtin_nontemporal_load(enc + ((size_t)t * Bv + b) * Hv + h);
    acc += w * e;
  }
  part[((size_t)chunk * Bv + b) * Hv + h] = acc;
}

__global__ void k_ctx_reduce(const float* __restrict__ part,
                             float* __restrict__ context) {
  int idx = blockIdx.x * blockDim.x + threadIdx.x;   // 16384
  float s = 0.f;
  for (int c = 0; c < 32; ++c) s += part[(size_t)c * Bv * Hv + idx];
  context[idx] = s;
}

// ---------------------------------------------------------------------------
// x = [motion, context] @ pre_W.T + pre_b
// ---------------------------------------------------------------------------
__global__ void k_pre(const float* __restrict__ motion,
                      const float* __restrict__ context,
                      const float* __restrict__ pre_W,   // [H][D+H]
                      const float* __restrict__ pre_b,
                      float* __restrict__ x) {
  int idx = blockIdx.x * blockDim.x + threadIdx.x;   // 16384
  int b = idx >> 8, g = idx & 255;
  const float* wr = pre_W + (size_t)g * (Dv + Hv);
  float acc = dot_n(motion + (size_t)b * Dv, wr, Dv);
  acc += dot_n(context + (size_t)b * Hv, wr + Dv, Hv);
  x[idx] = acc + pre_b[g];
}

// BatchNorm stats over batch (population var, ddof=0) + fused rsqrt
__global__ void k_bnstats(const float* __restrict__ x,
                          float* __restrict__ mu, float* __restrict__ rstd) {
  int g = threadIdx.x;                                // 1 block, 256 threads
  float s = 0.f, s2 = 0.f;
  for (int b = 0; b < Bv; ++b) {
    float val = x[(size_t)b * Hv + g];
    s += val; s2 += val * val;
  }
  float m = s * (1.0f / Bv);
  float var = s2 * (1.0f / Bv) - m * m;
  mu[g] = m;
  rstd[g] = rsqrtf(var + EPSv);
}

__global__ void k_bnrelu(const float* __restrict__ x,
                         const float* __restrict__ mu,
                         const float* __restrict__ rstd,
                         const float* __restrict__ gamma,
                         const float* __restrict__ beta,
                         float* __restrict__ xr) {
  int idx = blockIdx.x * blockDim.x + threadIdx.x;   // 16384
  int g = idx & 255;
  float val = (x[idx] - mu[g]) * rstd[g] * gamma[g] + beta[g];
  xr[idx] = fmaxf(val, 0.f);
}

// gi = xr @ W_ih.T + b_ih ; gh = h_prev @ W_hh.T + b_hh  (one kernel)
__global__ void k_gates(const float* __restrict__ xr,
                        const float* __restrict__ h_prev,
                        const float* __restrict__ W_ih,
                        const float* __restrict__ b_ih,
                        const float* __restrict__ W_hh,
                        const float* __restrict__ b_hh,
                        float* __restrict__ gi, float* __restrict__ gh) {
  int idx = blockIdx.x * blockDim.x + threadIdx.x;   // 98304
  int half = idx / (Bv * 3 * Hv);
  int o = idx - half * (Bv * 3 * Hv);
  int b = o / (3 * Hv), j = o - b * (3 * Hv);
  if (half == 0) {
    gi[o] = dot_n(xr + (size_t)b * Hv, W_ih + (size_t)j * Hv, Hv) + b_ih[j];
  } else {
    gh[o] = dot_n(h_prev + (size_t)b * Hv, W_hh + (size_t)j * Hv, Hv) + b_hh[j];
  }
}

// GRU combine -> h_new written straight into d_out hidden slot
__global__ void k_gru(const float* __restrict__ gi,
                      const float* __restrict__ gh,
                      const float* __restrict__ h_prev,
                      float* __restrict__ hidden_out) {
  int idx = blockIdx.x * blockDim.x + threadIdx.x;   // 16384
  int b = idx >> 8, j = idx & 255;
  const float* gib = gi + (size_t)b * 3 * Hv;
  const float* ghb = gh + (size_t)b * 3 * Hv;
  float r = sigmoidf_(gib[j] + ghb[j]);
  float z = sigmoidf_(gib[Hv + j] + ghb[Hv + j]);
  float n = tanhf(gib[2 * Hv + j] + r * ghb[2 * Hv + j]);
  float hp = h_prev[idx];
  hidden_out[idx] = (1.f - z) * n + z * hp;
}

__global__ void k_out(const float* __restrict__ h_new,
                      const float* __restrict__ out_W,  // [O][H]
                      const float* __restrict__ out_b,
                      float* __restrict__ out) {
  int idx = blockIdx.x * blockDim.x + threadIdx.x;   // 8192
  int b = idx >> 7, o = idx & 127;
  out[idx] = dot_n(h_new + (size_t)b * Hv, out_W + (size_t)o * Hv, Hv) + out_b[o];
}

// ---------------------------------------------------------------------------
// launcher
// ---------------------------------------------------------------------------
extern "C" void kernel_launch(void* const* d_in, const int* in_sizes, int n_in,
                              void* d_out, int out_size, void* d_ws, size_t ws_size,
                              hipStream_t stream) {
  const float* motion   = (const float*)d_in[0];
  const float* last_h   = (const float*)d_in[1];   // (1,B,H) -> [B][H]
  const float* enc      = (const float*)d_in[2];
  const float* attn_W   = (const float*)d_in[3];
  const float* attn_b   = (const float*)d_in[4];
  const float* v        = (const float*)d_in[5];
  const float* pre_W    = (const float*)d_in[6];
  const float* pre_b    = (const float*)d_in[7];
  const float* bn_gamma = (const float*)d_in[8];
  const float* bn_beta  = (const float*)d_in[9];
  const float* W_ih     = (const float*)d_in[10];
  const float* b_ih     = (const float*)d_in[11];
  const float* W_hh     = (const float*)d_in[12];
  const float* b_hh     = (const float*)d_in[13];
  const float* out_W    = (const float*)d_in[14];
  const float* out_b    = (const float*)d_in[15];

  float* out_output = (float*)d_out;                     // [B][O]
  float* out_hidden = (float*)d_out + Bv * Ov;           // [1][B][H]
  float* out_attn   = (float*)d_out + Bv * Ov + Bv * Hv; // [B][1][T]

  char* ws = (char*)d_ws;
  float*  hpT     = (float*)(ws + 0);          //  64 KB  [H][B]
  bf16_t* we      = (bf16_t*)(ws + 65536);     // 128 KB  [H][H] bf16
  float*  scores  = (float*)(ws + 196608);     //   1 MB  [B][T]
  float*  part    = (float*)(ws + 1245184);    //   2 MB  [32][B][H]
  float*  context = (float*)(ws + 3342336);    //  64 KB
  float*  x       = (float*)(ws + 3407872);    //  64 KB
  float*  mu      = (float*)(ws + 3473408);    //   1 KB
  float*  rstd    = (float*)(ws + 3474432);    //   1 KB
  float*  xr      = (float*)(ws + 3475456);    //  64 KB
  float*  gi      = (float*)(ws + 3540992);    // 192 KB
  float*  gh      = (float*)(ws + 3737600);    // 192 KB

  k_cvt_we  <<<256, 256, 0, stream>>>(attn_W, we);
  k_hidproj <<<64, 256, 0, stream>>>(last_h, attn_W, attn_b, hpT);
  k_scores  <<<NBLOCK, 256, 0, stream>>>(enc, we, hpT, v, scores);
  k_softmax <<<Bv, 256, 0, stream>>>(scores, out_attn);
  k_ctx_part<<<dim3(32, Bv), Hv, 0, stream>>>(enc, out_attn, part);
  k_ctx_reduce<<<64, 256, 0, stream>>>(part, context);
  k_pre     <<<64, 256, 0, stream>>>(motion, context, pre_W, pre_b, x);
  k_bnstats <<<1, 256, 0, stream>>>(x, mu, rstd);
  k_bnrelu  <<<64, 256, 0, stream>>>(x, mu, rstd, bn_gamma, bn_beta, xr);
  k_gates   <<<384, 256, 0, stream>>>(xr, last_h, W_ih, b_ih, W_hh, b_hh, gi, gh);
  k_gru     <<<64, 256, 0, stream>>>(gi, gh, last_h, out_hidden);
  k_out     <<<32, 256, 0, stream>>>(out_hidden, out_W, out_b, out_output);
}
